// ContextVectorLLM_16647293239421
// MI455X (gfx1250) — matrix operand order repaired and checked by reading, not compiled, not run on hardware
//
#include <hip/hip_runtime.h>
#include <hip/hip_bf16.h>
#include <stdint.h>
#include <stddef.h>

// ---------------------------------------------------------------------------
// ContextVectorLLM for MI455X (gfx1250, wave32, WMMA)
//   Phase 0: gather embeddings -> bf16 token tiles [S][16][512] (M padded 8->16)
//   Phase 0b: transpose-convert all weights fp32 [K][N] -> bf16 [N][K]
//   Phase 1: 16-WG cooperative recurrence (software grid barrier), WMMA bf16,
//            context in LDS, h_t archived to Hbuf (bf16), trajectory to d_out.
//            K-loops are 4-deep software-pipelined (single wave per SIMD has
//            no occupancy to hide L2 latency -> hide it with ILP).
//   Phase 2: logits = Hbuf[4096,1024] @ WoutT^T + bout, full-chip WMMA GEMM,
//            operands L2-resident (WoutT bf16 = 64MB, H = 8MB << 192MB L2),
//            2-deep pipelined K-loop.
// ---------------------------------------------------------------------------

typedef __attribute__((ext_vector_type(16))) __bf16 v16bf;
typedef __attribute__((ext_vector_type(8)))  __bf16 v8bf;
typedef __attribute__((ext_vector_type(8)))  float  v8f;

#define B_      8
#define S_      512
#define VOCAB_  32000
#define EMBED_  512
#define CTX_    256
#define HID_    1024
#define KIN_    768      // EMBED_ + CTX_
#define NWG     16       // cooperative workgroups in recurrence

// ---- WMMA fragment loaders (CDNA5 ISA 7.12.2 layouts, wave32) --------------

// A-matrix 16x32 bf16: lane L: m = L&15, h = L>>4.
//   V0..V3 = K[h*8 .. h*8+7], V4..V7 = K[16+h*8 .. 16+h*8+7]
__device__ __forceinline__ v16bf load_afrag(const __bf16* base, int ldk, int kbase) {
  const int lane = threadIdx.x & 31;
  const int m = lane & 15, hlf = (lane >> 4) & 1;
  const __bf16* p = base + (size_t)m * ldk + kbase + hlf * 8;
  union { v16bf v; v8bf h[2]; } u;
  u.h[0] = *(const v8bf*)(p);
  u.h[1] = *(const v8bf*)(p + 16);
  return u.v;
}

// B-matrix 32x16 bf16 from W^T stored [N][K]: lane L: n = L&15, h = L>>4.
//   V0..V3 = K[h*8 .. h*8+7], V4..V7 = K[16+h*8 .. 16+h*8+7] of column n
__device__ __forceinline__ v16bf load_bfrag(const __bf16* baseT, int ldk, int kbase) {
  const int lane = threadIdx.x & 31;
  const int n = lane & 15, hlf = (lane >> 4) & 1;
  const __bf16* p = baseT + (size_t)n * ldk + kbase + hlf * 8;
  union { v16bf v; v8bf h[2]; } u;
  u.h[0] = *(const v8bf*)(p);
  u.h[1] = *(const v8bf*)(p + 16);
  return u.v;
}

__device__ __forceinline__ v8f wmma_bf16(v16bf a, v16bf b, v8f c) {
  // (neg_a, A, neg_b, B, c_mod, C, reuse_a, reuse_b)
  return __builtin_amdgcn_wmma_f32_16x16x32_bf16(false, a, false, b, (short)0, c,
                                                 false, false);
}

// ---- software grid barrier (monotonic counter, deterministic) --------------
__device__ __forceinline__ void grid_barrier(unsigned* cnt, unsigned target) {
  __syncthreads();
  if (threadIdx.x == 0) {
    __threadfence();                       // release prior global writes
    atomicAdd(cnt, 1u);
    while (atomicAdd(cnt, 0u) < target) __builtin_amdgcn_s_sleep(1);
    __threadfence();                       // acquire peers' writes
  }
  __syncthreads();
}

// ---- Phase 0b: W fp32 [K][N] -> W^T bf16 [N][K] ----------------------------
__global__ void transpose_convert_kernel(const float* __restrict__ in,
                                         __bf16* __restrict__ out, int K, int N) {
  long long idx = (long long)blockIdx.x * blockDim.x + threadIdx.x;
  if (idx >= (long long)K * N) return;
  int k = (int)(idx / N);
  int n = (int)(idx % N);
  out[(size_t)n * K + k] = (__bf16)in[idx];
}

// ---- Phase 0: embedding gather, bf16, M padded to 16 -----------------------
__global__ void gather_tok_kernel(const int* __restrict__ ids,
                                  const float* __restrict__ emb,
                                  __bf16* __restrict__ tokbf) {
  int s = blockIdx.x;
  for (int i = threadIdx.x; i < 16 * EMBED_; i += blockDim.x) {
    int row = i >> 9, col = i & 511;
    float v = 0.f;
    if (row < B_) {
      int id = ids[(size_t)row * S_ + s];
      v = emb[(size_t)id * EMBED_ + col];
    }
    tokbf[((size_t)s * 16 + row) * EMBED_ + col] = (__bf16)v;
  }
}

// ---- Phase 1: cooperative recurrence ---------------------------------------
__global__ void __launch_bounds__(128) recurrence_kernel(
    const __bf16* __restrict__ tokbf,
    const __bf16* __restrict__ W0T, const __bf16* __restrict__ W1T,
    const __bf16* __restrict__ WcuT, const __bf16* __restrict__ WfT,
    const __bf16* __restrict__ WiT,
    const float* __restrict__ b0, const float* __restrict__ b1,
    const float* __restrict__ bcu, const float* __restrict__ bfg,
    const float* __restrict__ big,
    const float* __restrict__ lng, const float* __restrict__ lnb,
    __bf16* __restrict__ h1x, __bf16* __restrict__ h2x,
    float* __restrict__ ncpre, __bf16* __restrict__ Hbuf,
    float* __restrict__ traj, unsigned* __restrict__ cnt) {

  __shared__ __align__(16) float  ctx_f[16][CTX_];   // fp32 context (padded M=16)
  __shared__ __align__(16) __bf16 ctx_b[16][CTX_];   // bf16 mirror for WMMA A-frags
  __shared__ __align__(16) float  gate_lds[3][16][16];

  const int wg   = blockIdx.x;        // 0..15
  const int tid  = threadIdx.x;       // 0..127 (4 waves)
  const int wave = tid >> 5;
  const int lane = tid & 31;
  const int nloc = lane & 15;
  const int hlf  = (lane >> 4) & 1;

  for (int i = tid; i < 16 * CTX_; i += 128) {
    ctx_f[0][i] = 0.f;
    ctx_b[0][i] = (__bf16)0.f;
  }
  __syncthreads();

  unsigned bseq = 0;

  for (int s = 0; s < S_; ++s) {
    // emit pre-update context trajectory; WG owns cols [wg*16, wg*16+16)
    {
      int m = tid >> 4;                // 0..7 (real batch rows only)
      int j = tid & 15;
      int col = wg * 16 + j;
      traj[((size_t)m * S_ + s) * CTX_ + col] = ctx_f[m][col];
    }

    const __bf16* tok_s = tokbf + (size_t)s * 16 * EMBED_;

    // -- stage 1: h1 = relu([tok, ctx] @ W0 + b0); WG cols [wg*64, wg*64+64)
    //    4-deep pipelined K-loop (24 K-tiles)
    {
      int n0 = wg * 64 + wave * 16;
      const __bf16* Wrow = W0T + (size_t)n0 * KIN_;
      v8f acc = {0, 0, 0, 0, 0, 0, 0, 0};
      v16bf a[4], bw[4];
      auto ldA = [&](int kt) -> v16bf {
        int kb = kt * 32;
        return (kb < EMBED_) ? load_afrag(tok_s, EMBED_, kb)
                             : load_afrag(&ctx_b[0][0], CTX_, kb - EMBED_);
      };
      #pragma unroll
      for (int p = 0; p < 3; ++p) { a[p] = ldA(p); bw[p] = load_bfrag(Wrow, KIN_, p * 32); }
      #pragma unroll
      for (int kt = 0; kt < KIN_ / 32; ++kt) {
        int cur = kt & 3, nxt = (kt + 3) & 3;
        if (kt + 3 < KIN_ / 32) { a[nxt] = ldA(kt + 3); bw[nxt] = load_bfrag(Wrow, KIN_, (kt + 3) * 32); }
        acc = wmma_bf16(a[cur], bw[cur], acc);
      }
      int n = n0 + nloc;
      float bias = b0[n];
      #pragma unroll
      for (int r = 0; r < 8; ++r) {
        float v = acc[r] + bias;
        v = v > 0.f ? v : 0.f;
        int m = r + hlf * 8;
        h1x[(size_t)m * HID_ + n] = (__bf16)v;
      }
    }
    grid_barrier(cnt, (++bseq) * NWG);

    // -- stage 2: h2 = relu(h1 @ W1 + b1); archive rows 0..7 into Hbuf
    {
      int n0 = wg * 64 + wave * 16;
      const __bf16* Wrow = W1T + (size_t)n0 * HID_;
      v8f acc = {0, 0, 0, 0, 0, 0, 0, 0};
      v16bf a[4], bw[4];
      #pragma unroll
      for (int p = 0; p < 3; ++p) { a[p] = load_afrag(h1x, HID_, p * 32); bw[p] = load_bfrag(Wrow, HID_, p * 32); }
      #pragma unroll
      for (int kt = 0; kt < HID_ / 32; ++kt) {
        int cur = kt & 3, nxt = (kt + 3) & 3;
        if (kt + 3 < HID_ / 32) {
          a[nxt]  = load_afrag(h1x, HID_, (kt + 3) * 32);
          bw[nxt] = load_bfrag(Wrow, HID_, (kt + 3) * 32);
        }
        acc = wmma_bf16(a[cur], bw[cur], acc);
      }
      int n = n0 + nloc;
      float bias = b1[n];
      #pragma unroll
      for (int r = 0; r < 8; ++r) {
        float v = acc[r] + bias;
        v = v > 0.f ? v : 0.f;
        int m = r + hlf * 8;
        __bf16 hv = (__bf16)v;
        h2x[(size_t)m * HID_ + n] = hv;
        if (m < B_) Hbuf[((size_t)m * S_ + s) * HID_ + n] = hv;  // row = b*S + s
      }
    }
    grid_barrier(cnt, (++bseq) * NWG);

    // -- stage 3: gates; wave 0/1/2 -> delta/f/i, cols [wg*16, wg*16+16)
    if (wave < 3) {
      const __bf16* WT = (wave == 0) ? WcuT : (wave == 1) ? WfT : WiT;
      const float*  bb = (wave == 0) ? bcu  : (wave == 1) ? bfg : big;
      int n0 = wg * 16;
      const __bf16* Wrow = WT + (size_t)n0 * HID_;
      v8f acc = {0, 0, 0, 0, 0, 0, 0, 0};
      v16bf a[4], bw[4];
      #pragma unroll
      for (int p = 0; p < 3; ++p) { a[p] = load_afrag(h2x, HID_, p * 32); bw[p] = load_bfrag(Wrow, HID_, p * 32); }
      #pragma unroll
      for (int kt = 0; kt < HID_ / 32; ++kt) {
        int cur = kt & 3, nxt = (kt + 3) & 3;
        if (kt + 3 < HID_ / 32) {
          a[nxt]  = load_afrag(h2x, HID_, (kt + 3) * 32);
          bw[nxt] = load_bfrag(Wrow, HID_, (kt + 3) * 32);
        }
        acc = wmma_bf16(a[cur], bw[cur], acc);
      }
      float bias = bb[n0 + nloc];
      #pragma unroll
      for (int r = 0; r < 8; ++r) {
        float v = acc[r] + bias;
        if (wave > 0) v = 1.f / (1.f + __expf(-v));   // sigmoid for f,i
        gate_lds[wave][r + hlf * 8][nloc] = v;
      }
    }
    __syncthreads();

    // new_c_pre = f*ctx + i*delta for this WG's 16 columns -> global exchange
    for (int i = tid; i < 256; i += 128) {
      int m = i >> 4, j = i & 15;
      int col = wg * 16 + j;
      float d  = gate_lds[0][m][j];
      float fg = gate_lds[1][m][j];
      float ig = gate_lds[2][m][j];
      ncpre[(size_t)m * CTX_ + col] = fg * ctx_f[m][col] + ig * d;
    }
    grid_barrier(cnt, (++bseq) * NWG);

    // -- LayerNorm + clip (redundant per WG; avoids a 4th barrier)
    {
      int row = tid >> 3;     // 0..15
      int j   = tid & 7;      // 8 threads per row
      float s1 = 0.f, s2 = 0.f;
      for (int c = 0; c < 32; ++c) {
        float x = ncpre[(size_t)row * CTX_ + c * 8 + j];
        s1 += x; s2 += x * x;
      }
      #pragma unroll
      for (int msk = 1; msk < 8; msk <<= 1) {
        s1 += __shfl_xor(s1, msk, 32);
        s2 += __shfl_xor(s2, msk, 32);
      }
      float mean = s1 * (1.f / CTX_);
      float var  = s2 * (1.f / CTX_) - mean * mean;
      float rstd = rsqrtf(var + 1e-5f);
      for (int c = 0; c < 32; ++c) {
        int col = c * 8 + j;
        float x = ncpre[(size_t)row * CTX_ + col];
        float v = (x - mean) * rstd * lng[col] + lnb[col];
        v = fminf(10.f, fmaxf(-10.f, v));
        ctx_f[row][col] = v;
        ctx_b[row][col] = (__bf16)v;
      }
    }
    __syncthreads();
  }
}

// ---- Phase 2: logits GEMM, 128x128 block tile, 64x32 wave tile,
//      2-deep pipelined K-loop ------------------------------------------------
__global__ void __launch_bounds__(256) logits_gemm_kernel(
    const __bf16* __restrict__ Hbuf, const __bf16* __restrict__ WoutT,
    const float* __restrict__ bout, float* __restrict__ out) {
  const int w    = threadIdx.x >> 5;
  const int lane = threadIdx.x & 31;
  const int nloc = lane & 15, hlf = (lane >> 4) & 1;
  const int mbase = blockIdx.y * 128 + (w & 1) * 64;
  const int nbase = blockIdx.x * 128 + (w >> 1) * 32;

  v8f acc[4][2];
  #pragma unroll
  for (int f = 0; f < 4; ++f)
    #pragma unroll
    for (int g = 0; g < 2; ++g) acc[f][g] = (v8f){0, 0, 0, 0, 0, 0, 0, 0};

  v16bf abuf[2][4], bbuf[2][2];
  auto ldA = [&](int buf, int kb) {
    #pragma unroll
    for (int f = 0; f < 4; ++f)
      abuf[buf][f] = load_afrag(Hbuf + (size_t)(mbase + f * 16) * HID_, HID_, kb);
  };
  auto ldB = [&](int buf, int kb) {
    #pragma unroll
    for (int g = 0; g < 2; ++g)
      bbuf[buf][g] = load_bfrag(WoutT + (size_t)(nbase + g * 16) * HID_, HID_, kb);
  };

  ldA(0, 0);
  ldB(0, 0);
  #pragma unroll
  for (int kt = 0; kt < HID_ / 32; ++kt) {
    int cur = kt & 1, nxt = cur ^ 1;
    if (kt + 1 < HID_ / 32) {
      // hint the K-slice after next into the cache hierarchy
      if (kt + 2 < HID_ / 32)
        __builtin_prefetch((const void*)(WoutT + (size_t)(nbase + nloc) * HID_ + (kt + 2) * 32), 0, 1);
      ldA(nxt, (kt + 1) * 32);
      ldB(nxt, (kt + 1) * 32);
    }
    #pragma unroll
    for (int f = 0; f < 4; ++f)
      #pragma unroll
      for (int g = 0; g < 2; ++g)
        acc[f][g] = wmma_bf16(abuf[cur][f], bbuf[cur][g], acc[f][g]);
  }

  #pragma unroll
  for (int g = 0; g < 2; ++g) {
    int n = nbase + g * 16 + nloc;
    float bias = bout[n];
    #pragma unroll
    for (int f = 0; f < 4; ++f) {
      int mrow = mbase + f * 16 + hlf * 8;
      #pragma unroll
      for (int r = 0; r < 8; ++r)
        out[(size_t)(mrow + r) * VOCAB_ + n] = acc[f][g][r] + bias;
    }
  }
}

// ---------------------------------------------------------------------------
extern "C" void kernel_launch(void* const* d_in, const int* in_sizes, int n_in,
                              void* d_out, int out_size, void* d_ws, size_t ws_size,
                              hipStream_t stream) {
  (void)in_sizes; (void)n_in; (void)out_size; (void)ws_size;

  const int*   ids  = (const int*)  d_in[0];
  const float* emb  = (const float*)d_in[1];
  const float* W0   = (const float*)d_in[2];
  const float* b0   = (const float*)d_in[3];
  const float* W1   = (const float*)d_in[4];
  const float* b1   = (const float*)d_in[5];
  const float* Wout = (const float*)d_in[6];
  const float* bout = (const float*)d_in[7];
  const float* Wcu  = (const float*)d_in[8];
  const float* bcu  = (const float*)d_in[9];
  const float* Wf   = (const float*)d_in[10];
  const float* bf   = (const float*)d_in[11];
  const float* Wi   = (const float*)d_in[12];
  const float* bi   = (const float*)d_in[13];
  const float* lng  = (const float*)d_in[14];
  const float* lnb  = (const float*)d_in[15];

  char* ws = (char*)d_ws;
  size_t off = 0;
  auto alloc = [&](size_t bytes) -> char* {
    char* p = ws + off;
    off += (bytes + 255) & ~(size_t)255;
    return p;
  };

  unsigned* bar   = (unsigned*)alloc(256);
  __bf16*   tokbf = (__bf16*)alloc((size_t)S_ * 16 * EMBED_ * 2);   //  8 MB
  __bf16*   W0T   = (__bf16*)alloc((size_t)HID_ * KIN_ * 2);        //  1.5 MB
  __bf16*   W1T   = (__bf16*)alloc((size_t)HID_ * HID_ * 2);        //  2 MB
  __bf16*   WcuT  = (__bf16*)alloc((size_t)CTX_ * HID_ * 2);
  __bf16*   WfT   = (__bf16*)alloc((size_t)CTX_ * HID_ * 2);
  __bf16*   WiT   = (__bf16*)alloc((size_t)CTX_ * HID_ * 2);
  __bf16*   WoutT = (__bf16*)alloc((size_t)VOCAB_ * HID_ * 2);      // 62.5 MB (L2-resident)
  __bf16*   h1x   = (__bf16*)alloc((size_t)16 * HID_ * 2);
  __bf16*   h2x   = (__bf16*)alloc((size_t)16 * HID_ * 2);
  float*    ncpre = (float*) alloc((size_t)16 * CTX_ * 4);
  __bf16*   Hbuf  = (__bf16*)alloc((size_t)B_ * S_ * HID_ * 2);     //  8 MB

  float* logits = (float*)d_out;
  float* traj   = logits + (size_t)B_ * S_ * VOCAB_;

  hipMemsetAsync(bar, 0, 256, stream);   // grid-barrier counter

  auto tlaunch = [&](const float* in, __bf16* outp, int K, int N) {
    long long tot = (long long)K * N;
    int blocks = (int)((tot + 255) / 256);
    transpose_convert_kernel<<<blocks, 256, 0, stream>>>(in, outp, K, N);
  };
  tlaunch(W0,   W0T,   KIN_, HID_);
  tlaunch(W1,   W1T,   HID_, HID_);
  tlaunch(Wcu,  WcuT,  HID_, CTX_);
  tlaunch(Wf,   WfT,   HID_, CTX_);
  tlaunch(Wi,   WiT,   HID_, CTX_);
  tlaunch(Wout, WoutT, HID_, VOCAB_);

  gather_tok_kernel<<<S_, 256, 0, stream>>>(ids, emb, tokbf);

  recurrence_kernel<<<NWG, 128, 0, stream>>>(
      tokbf, W0T, W1T, WcuT, WfT, WiT,
      b0, b1, bcu, bf, bi, lng, lnb,
      h1x, h2x, ncpre, Hbuf, traj, bar);

  logits_gemm_kernel<<<dim3(VOCAB_ / 128, (B_ * S_) / 128), 256, 0, stream>>>(
      Hbuf, WoutT, bout, logits);
}
